// Seq_Model_34540126994788
// MI455X (gfx1250) — matrix operand order repaired and checked by reading, not compiled, not run on hardware
//
#include <hip/hip_runtime.h>
#include <hip/hip_bf16.h>

// ---------------------------------------------------------------------------
// Problem constants (match reference setup_inputs)
// ---------------------------------------------------------------------------
constexpr int kB = 512;   // batch
constexpr int kT = 512;   // time steps
constexpr int kD = 128;   // input features

typedef __attribute__((ext_vector_type(16))) __bf16 v16bf;
typedef __attribute__((ext_vector_type(8)))  __bf16 v8bf;
typedef __attribute__((ext_vector_type(4)))  __bf16 v4bf;
typedef __attribute__((ext_vector_type(8)))  float  v8f;

__device__ __forceinline__ v8f wmma_bf16(v16bf a, v16bf b, v8f c) {
  // v_wmma_f32_16x16x32_bf16  (A,B bf16, C/D f32)
  return __builtin_amdgcn_wmma_f32_16x16x32_bf16(
      /*neg_a=*/false, a, /*neg_b=*/false, b,
      /*c_mod=*/(short)0, c, /*reuse_a=*/false, /*reuse_b=*/false);
}

__device__ __forceinline__ float sigmoidf_fast(float x) {
  return 1.0f / (1.0f + __expf(-x));
}

// ---------------------------------------------------------------------------
// f32 -> bf16 conversion, 4 elements/thread (float4 in, 8-byte b64 out)
// n4 = element count / 4 (all our sizes are multiples of 4)
// ---------------------------------------------------------------------------
__global__ void cvt_bf16x4_kernel(v4bf* __restrict__ dst,
                                  const float4* __restrict__ src, int n4) {
  int i = blockIdx.x * blockDim.x + threadIdx.x;
  int stride = gridDim.x * blockDim.x;
  for (; i < n4; i += stride) {
    float4 v = src[i];
    v4bf o;
    o[0] = (__bf16)v.x; o[1] = (__bf16)v.y;
    o[2] = (__bf16)v.z; o[3] = (__bf16)v.w;
    dst[i] = o;
  }
}

// combined bias = b_ih + b_hh, fwd rows then bwd rows
__global__ void bias_pack_kernel(float* __restrict__ dst,
                                 const float* __restrict__ bihf,
                                 const float* __restrict__ bhhf,
                                 const float* __restrict__ bihb,
                                 const float* __restrict__ bhhb, int fourH) {
  int i = blockIdx.x * blockDim.x + threadIdx.x;
  if (i < fourH)            dst[i] = bihf[i] + bhhf[i];
  else if (i < 2 * fourH)   dst[i] = bihb[i - fourH] + bhhb[i - fourH];
}

// ---------------------------------------------------------------------------
// Input-projection GEMM:
//   gx[m, n] = sum_d A[m, d] * W[n, d] + bias[n]      (f32 out, bf16 A/W)
// Logical row m = t*kB + b.  A element (m, d) lives at b*sA_b + t*sA_t + d,
// covering x [B,T,D] (sA_b=T*D, sA_t=D) and hidden buffers [T,B,C]
// (sA_b=C, sA_t=B*C).
//
// Output is stored TILED in WMMA per-lane register order:
//   out[(mtile*(N/16) + ntile)*256 + lane*8 + r]  (mtile == blockIdx.x)
// The scan kernel reads the same order back, so the C-operand layout
// round-trips without any scatter/gather: 2x b128 stores here, 2x b128
// loads there, fully coalesced.
// ---------------------------------------------------------------------------
__global__ void gx_gemm_kernel(const __bf16* __restrict__ A,
                               const __bf16* __restrict__ W,
                               const float*  __restrict__ bias,
                               float* __restrict__ out,
                               int sA_b, int sA_t, int Din, int N) {
  const int tid  = threadIdx.x;
  const int wave = tid >> 5;
  const int lane = tid & 31;
  const int row  = lane & 15;   // M row within tile (A), N col (B)
  const int hi   = lane >> 4;   // lane half selects K sub-range

  const int m0 = blockIdx.x * 16;
  const int n0 = blockIdx.y * 64 + wave * 16;

  // A row pointer for this lane's M row
  const int m = m0 + row;
  const int b = m & (kB - 1);
  const int t = m >> 9;          // m / kB  (kB == 512)
  const __bf16* __restrict__ arow = A + (size_t)b * sA_b + (size_t)t * sA_t;
  const __bf16* __restrict__ wrow = W + (size_t)(n0 + row) * Din;

  v8f acc;
  const float bv = bias[n0 + row];
#pragma unroll
  for (int r = 0; r < 8; ++r) acc[r] = bv;

  for (int k0 = 0; k0 < Din; k0 += 32) {
    // A fragment: K = k0 + e + hi*16, one contiguous run of 16
    v16bf a = *(const v16bf*)&arow[k0 + hi * 16];
    // B fragment: K = k0 + e + hi*16, one contiguous run of 16
    v16bf bm = *(const v16bf*)&wrow[k0 + hi * 16];

    acc = wmma_bf16(a, bm, acc);
  }

  // tiled per-lane store (register order): 2x global_store_b128
  float* __restrict__ op =
      out + ((size_t)blockIdx.x * (N >> 4) + (blockIdx.y * 4 + wave)) * 256 +
      lane * 8;
  *(v8f*)op = acc;
}

// ---------------------------------------------------------------------------
// Recurrent scan, one direction per blockIdx.y (0=fwd, 1=bwd).
// One workgroup = 16 batches, NW = H/4 waves; wave w owns gate columns
// [16w, 16w+16).  w_hh B-fragment stays in VGPRs for the whole T loop.
// Per step:
//   acc = WMMA(h_prev (LDS bf16, K zero-padded to 32), w_hh, gx_tile)
//   (next step's gx tile load is issued immediately after -> double buffer)
//   gates -> LDS (v8f, per-lane order), barrier, elementwise LSTM cell,
//   new h -> LDS + global, barrier.
// gx: tiled layout from gx_gemm_kernel, tile index t*(kB/16)+bg, column
//     tile cb = dir*(H/4) + wave.
// hout: [T, B, 2H] bf16 (fwd h cols 0..H-1, bwd H..2H-1)
// ---------------------------------------------------------------------------
template <int H>
__global__ void lstm_scan_kernel(const float* __restrict__ gx,
                                 const __bf16* __restrict__ whh_f,
                                 const __bf16* __restrict__ whh_b,
                                 __bf16* __restrict__ hout) {
  constexpr int NW = H / 4;      // waves per block
  constexpr int CT = H / 2;      // column tiles per gx row-tile (8H/16)
  constexpr int NT = 32 * NW;    // threads per block

  const int tid  = threadIdx.x;
  const int wave = tid >> 5;
  const int lane = tid & 31;
  const int row  = lane & 15;
  const int hi   = lane >> 4;

  const int bg  = blockIdx.x;        // batch group (16 batches)
  const int b0  = bg * 16;
  const int dir = blockIdx.y;
  const __bf16* __restrict__ whh = dir ? whh_b : whh_f;

  __shared__ __align__(16) __bf16 hsh[16][32];   // h_{t-1}, zero-padded K
  __shared__ __align__(32) float  gsh[NT * 8];   // gates, per-lane order

  for (int i = tid; i < 16 * 32; i += NT) ((__bf16*)hsh)[i] = (__bf16)0.0f;

  // Resident w_hh B-fragment: Bmat[k, n] = whh[n*H + k]; K = e + hi*16
  v16bf bm;
  {
    const int n = wave * 16 + row;
#pragma unroll
    for (int e = 0; e < 16; ++e) {
      const int K = e + hi * 16;
      bm[e] = (K < H) ? whh[n * H + K] : (__bf16)0.0f;
    }
  }

  // Cell state: 16*H values, 2 per thread
  float c0 = 0.0f, c1 = 0.0f;
  const int idx0 = tid, idx1 = tid + NT;
  const int m_0 = idx0 / H, k_0 = idx0 % H;
  const int m_1 = idx1 / H, k_1 = idx1 % H;

  // gate flat-LDS index for gate element (m, col):
  //   lane_g = (col/16)*32 + (m/8)*16 + (col%16), slot = m%8
  auto gidx = [](int mm, int col) {
    return ((col >> 4) * 32 + (mm >> 3) * 16 + (col & 15)) * 8 + (mm & 7);
  };

  const int cb = dir * (H / 4) + wave;  // this wave's gx column tile
  auto cbase = [&](int t) -> const float* {
    return gx + (((size_t)t * (kB / 16) + bg) * CT + cb) * 256 + lane * 8;
  };

  __syncthreads();

  // Prime the C-operand double buffer with step 0's gx tile
  v8f ccur = *(const v8f*)cbase(dir ? (kT - 1) : 0);

  for (int step = 0; step < kT; ++step) {
    const int t = dir ? (kT - 1 - step) : step;

    // A fragment from previous h (LDS): 2x ds_load_b128
    v8bf alo = *(const v8bf*)&hsh[row][hi * 16];
    v8bf ahi = *(const v8bf*)&hsh[row][hi * 16 + 8];
    v16bf a = __builtin_shufflevector(alo, ahi, 0, 1, 2, 3, 4, 5, 6, 7,
                                      8, 9, 10, 11, 12, 13, 14, 15);

    v8f acc = wmma_bf16(a, bm, ccur);

    // Issue next step's gx tile load; it completes under the gate math
    if (step + 1 < kT) {
      const int tn = dir ? (kT - 2 - step) : (step + 1);
      ccur = *(const v8f*)cbase(tn);
    }

    // gates -> LDS in per-lane order: 2x ds_store_b128
    *(v8f*)&gsh[(wave * 32 + lane) * 8] = acc;
    __syncthreads();

    // Elementwise LSTM cell update (2 (m,k) pairs per thread)
    __bf16* __restrict__ hrow =
        hout + ((size_t)t * kB + b0) * (2 * H) + dir * H;
    {
      float gi = gsh[gidx(m_0, k_0)],     gf = gsh[gidx(m_0, H + k_0)];
      float gg = gsh[gidx(m_0, 2*H+k_0)], go = gsh[gidx(m_0, 3*H+k_0)];
      c0 = sigmoidf_fast(gf) * c0 + sigmoidf_fast(gi) * tanhf(gg);
      const float h = sigmoidf_fast(go) * tanhf(c0);
      hsh[m_0][k_0] = (__bf16)h;
      hrow[m_0 * 2 * H + k_0] = (__bf16)h;
    }
    {
      float gi = gsh[gidx(m_1, k_1)],     gf = gsh[gidx(m_1, H + k_1)];
      float gg = gsh[gidx(m_1, 2*H+k_1)], go = gsh[gidx(m_1, 3*H+k_1)];
      c1 = sigmoidf_fast(gf) * c1 + sigmoidf_fast(gi) * tanhf(gg);
      const float h = sigmoidf_fast(go) * tanhf(c1);
      hsh[m_1][k_1] = (__bf16)h;
      hrow[m_1 * 2 * H + k_1] = (__bf16)h;
    }
    __syncthreads();
  }
}

// ---------------------------------------------------------------------------
// FC (16 -> 6) + softmax over the TIME axis, one block per batch element.
// h3: [T, B, 16] bf16.   out: [B, T, 6] f32.
// ---------------------------------------------------------------------------
__global__ void fc_softmax_kernel(const __bf16* __restrict__ h3,
                                  const float* __restrict__ fcw,  // [6,16]
                                  const float* __restrict__ fcb,  // [6]
                                  float* __restrict__ out) {
  const int b   = blockIdx.x;
  const int tid = threadIdx.x;           // 128 threads
  constexpr int NTH = 128;

  __shared__ float lg[kT * 6];           // logits for this batch, 12 KB
  __shared__ float red[NTH * 6];
  __shared__ float wsh[6 * 16 + 6];

  for (int i = tid; i < 6 * 16; i += NTH) wsh[i] = fcw[i];
  if (tid < 6) wsh[6 * 16 + tid] = fcb[tid];
  __syncthreads();

  // logits
  for (int t = tid; t < kT; t += NTH) {
    const v8bf* __restrict__ hp = (const v8bf*)(h3 + ((size_t)t * kB + b) * 16);
    v8bf h0 = hp[0], h1v = hp[1];
    float x[16];
#pragma unroll
    for (int j = 0; j < 8; ++j) { x[j] = (float)h0[j]; x[j + 8] = (float)h1v[j]; }
#pragma unroll
    for (int c = 0; c < 6; ++c) {
      float s = wsh[6 * 16 + c];
#pragma unroll
      for (int j = 0; j < 16; ++j) s += x[j] * wsh[c * 16 + j];
      lg[t * 6 + c] = s;
    }
  }
  __syncthreads();

  // max over T (per class)
  float mx[6];
#pragma unroll
  for (int c = 0; c < 6; ++c) mx[c] = -3.0e38f;
  for (int t = tid; t < kT; t += NTH)
#pragma unroll
    for (int c = 0; c < 6; ++c) mx[c] = fmaxf(mx[c], lg[t * 6 + c]);
#pragma unroll
  for (int c = 0; c < 6; ++c) red[tid * 6 + c] = mx[c];
  __syncthreads();
  for (int s = NTH / 2; s > 0; s >>= 1) {
    if (tid < s)
#pragma unroll
      for (int c = 0; c < 6; ++c)
        red[tid * 6 + c] = fmaxf(red[tid * 6 + c], red[(tid + s) * 6 + c]);
    __syncthreads();
  }
#pragma unroll
  for (int c = 0; c < 6; ++c) mx[c] = red[c];
  __syncthreads();

  // sum of exp over T (per class)
  float sm[6];
#pragma unroll
  for (int c = 0; c < 6; ++c) sm[c] = 0.0f;
  for (int t = tid; t < kT; t += NTH)
#pragma unroll
    for (int c = 0; c < 6; ++c) sm[c] += __expf(lg[t * 6 + c] - mx[c]);
#pragma unroll
  for (int c = 0; c < 6; ++c) red[tid * 6 + c] = sm[c];
  __syncthreads();
  for (int s = NTH / 2; s > 0; s >>= 1) {
    if (tid < s)
#pragma unroll
      for (int c = 0; c < 6; ++c) red[tid * 6 + c] += red[(tid + s) * 6 + c];
    __syncthreads();
  }
#pragma unroll
  for (int c = 0; c < 6; ++c) sm[c] = 1.0f / red[c];
  __syncthreads();

  // write softmax over time axis: out[b, t, c]
  for (int t = tid; t < kT; t += NTH)
#pragma unroll
    for (int c = 0; c < 6; ++c)
      out[((size_t)b * kT + t) * 6 + c] = __expf(lg[t * 6 + c] - mx[c]) * sm[c];
}

// ---------------------------------------------------------------------------
// Host driver
// ---------------------------------------------------------------------------
// Workspace layout (bytes):
//   R0: 67 MB  — x as bf16, later reused for h2 [T,B,32] bf16
//   R1: 256 MB — gx buffers (layer1 f32 tiled [T*B/16, 16, 256] is largest)
//   R2: 32 MB  — h1 [T,B,64] bf16, later reused for h3 [T,B,16] bf16
//   W:  ~110 KB — packed bf16 weights + combined f32 biases
static constexpr size_t OFF_R0 = 0;
static constexpr size_t SZ_R0  = (size_t)kB * kT * kD * 2;            // 67,108,864
static constexpr size_t OFF_R1 = OFF_R0 + SZ_R0;
static constexpr size_t SZ_R1  = (size_t)kT * kB * 256 * 4;           // 268,435,456
static constexpr size_t OFF_R2 = OFF_R1 + SZ_R1;
static constexpr size_t SZ_R2  = (size_t)kT * kB * 64 * 2;            // 33,554,432
static constexpr size_t OFF_W  = OFF_R2 + SZ_R2;

extern "C" void kernel_launch(void* const* d_in, const int* in_sizes, int n_in,
                              void* d_out, int out_size, void* d_ws,
                              size_t ws_size, hipStream_t stream) {
  (void)in_sizes; (void)n_in; (void)out_size; (void)ws_size;

  const float* x = (const float*)d_in[0];
  const float* w_ih1_f = (const float*)d_in[1];
  const float* w_hh1_f = (const float*)d_in[2];
  const float* b_ih1_f = (const float*)d_in[3];
  const float* b_hh1_f = (const float*)d_in[4];
  const float* w_ih1_b = (const float*)d_in[5];
  const float* w_hh1_b = (const float*)d_in[6];
  const float* b_ih1_b = (const float*)d_in[7];
  const float* b_hh1_b = (const float*)d_in[8];
  const float* w_ih2_f = (const float*)d_in[9];
  const float* w_hh2_f = (const float*)d_in[10];
  const float* b_ih2_f = (const float*)d_in[11];
  const float* b_hh2_f = (const float*)d_in[12];
  const float* w_ih2_b = (const float*)d_in[13];
  const float* w_hh2_b = (const float*)d_in[14];
  const float* b_ih2_b = (const float*)d_in[15];
  const float* b_hh2_b = (const float*)d_in[16];
  const float* w_ih3_f = (const float*)d_in[17];
  const float* w_hh3_f = (const float*)d_in[18];
  const float* b_ih3_f = (const float*)d_in[19];
  const float* b_hh3_f = (const float*)d_in[20];
  const float* w_ih3_b = (const float*)d_in[21];
  const float* w_hh3_b = (const float*)d_in[22];
  const float* b_ih3_b = (const float*)d_in[23];
  const float* b_hh3_b = (const float*)d_in[24];
  const float* fc_w    = (const float*)d_in[25];
  const float* fc_b    = (const float*)d_in[26];

  char* ws = (char*)d_ws;
  __bf16* xbf = (__bf16*)(ws + OFF_R0);
  float*  gxb = (float*)(ws + OFF_R1);
  __bf16* h1  = (__bf16*)(ws + OFF_R2);   // [T,B,64]
  __bf16* h2  = (__bf16*)(ws + OFF_R0);   // [T,B,32] (x dead by then)
  __bf16* h3  = (__bf16*)(ws + OFF_R2);   // [T,B,16] (h1 dead by then)

  size_t wo = OFF_W;
  __bf16* W1   = (__bf16*)(ws + wo); wo += 256 * 128 * 2;
  __bf16* W2   = (__bf16*)(ws + wo); wo += 128 * 64 * 2;
  __bf16* W3   = (__bf16*)(ws + wo); wo += 64 * 32 * 2;
  __bf16* WH1F = (__bf16*)(ws + wo); wo += 128 * 32 * 2;
  __bf16* WH1B = (__bf16*)(ws + wo); wo += 128 * 32 * 2;
  __bf16* WH2F = (__bf16*)(ws + wo); wo += 64 * 16 * 2;
  __bf16* WH2B = (__bf16*)(ws + wo); wo += 64 * 16 * 2;
  __bf16* WH3F = (__bf16*)(ws + wo); wo += 32 * 8 * 2;
  __bf16* WH3B = (__bf16*)(ws + wo); wo += 32 * 8 * 2;
  float*  BZ1  = (float*)(ws + wo);  wo += 256 * 4;
  float*  BZ2  = (float*)(ws + wo);  wo += 128 * 4;
  float*  BZ3  = (float*)(ws + wo);  wo += 64 * 4;

  auto cvt = [&](__bf16* dst, const float* src, int n) {
    int n4 = n / 4;
    int blocks = (n4 + 255) / 256;
    if (blocks > 4096) blocks = 4096;
    cvt_bf16x4_kernel<<<blocks, 256, 0, stream>>>((v4bf*)dst,
                                                  (const float4*)src, n4);
  };

  // --- weight / bias prep (tiny) ---
  cvt(W1,            w_ih1_f, 128 * 128);
  cvt(W1 + 128*128,  w_ih1_b, 128 * 128);
  cvt(W2,            w_ih2_f, 64 * 64);
  cvt(W2 + 64*64,    w_ih2_b, 64 * 64);
  cvt(W3,            w_ih3_f, 32 * 32);
  cvt(W3 + 32*32,    w_ih3_b, 32 * 32);
  cvt(WH1F, w_hh1_f, 128 * 32);
  cvt(WH1B, w_hh1_b, 128 * 32);
  cvt(WH2F, w_hh2_f, 64 * 16);
  cvt(WH2B, w_hh2_b, 64 * 16);
  cvt(WH3F, w_hh3_f, 32 * 8);
  cvt(WH3B, w_hh3_b, 32 * 8);
  bias_pack_kernel<<<1, 256, 0, stream>>>(BZ1, b_ih1_f, b_hh1_f, b_ih1_b, b_hh1_b, 128);
  bias_pack_kernel<<<1, 128, 0, stream>>>(BZ2, b_ih2_f, b_hh2_f, b_ih2_b, b_hh2_b, 64);
  bias_pack_kernel<<<1, 64, 0, stream>>>(BZ3, b_ih3_f, b_hh3_f, b_ih3_b, b_hh3_b, 32);

  // --- x -> bf16 ---
  cvt(xbf, x, kB * kT * kD);

  const int MT = kT * kB / 16;  // 16384 M-tiles

  // --- layer 1 ---
  gx_gemm_kernel<<<dim3(MT, 256 / 64), 128, 0, stream>>>(
      xbf, W1, BZ1, gxb, /*sA_b=*/kT * kD, /*sA_t=*/kD, /*Din=*/kD, /*N=*/256);
  lstm_scan_kernel<32><<<dim3(kB / 16, 2), 256, 0, stream>>>(gxb, WH1F, WH1B, h1);

  // --- layer 2 (input h1 [T,B,64]) ---
  gx_gemm_kernel<<<dim3(MT, 128 / 64), 128, 0, stream>>>(
      h1, W2, BZ2, gxb, /*sA_b=*/64, /*sA_t=*/kB * 64, /*Din=*/64, /*N=*/128);
  lstm_scan_kernel<16><<<dim3(kB / 16, 2), 128, 0, stream>>>(gxb, WH2F, WH2B, h2);

  // --- layer 3 (input h2 [T,B,32]) ---
  gx_gemm_kernel<<<dim3(MT, 1), 128, 0, stream>>>(
      h2, W3, BZ3, gxb, /*sA_b=*/32, /*sA_t=*/kB * 32, /*Din=*/32, /*N=*/64);
  lstm_scan_kernel<8><<<dim3(kB / 16, 2), 64, 0, stream>>>(gxb, WH3F, WH3B, h3);

  // --- FC + softmax over time ---
  fc_softmax_kernel<<<kB, 128, 0, stream>>>(h3, fc_w, fc_b, (float*)d_out);
}
